// GMN_match_hinge_70574902608348
// MI455X (gfx1250) — compile-verified, matrix operand order checked by reading, not run on hardware
//
#include <hip/hip_runtime.h>

// ---------------------------------------------------------------------------
// GMN match-hinge for MI455X (gfx1250, wave32, WMMA 16x16x32 f16 -> f32)
// ---------------------------------------------------------------------------

typedef __attribute__((ext_vector_type(16))) _Float16 v16h;
typedef __attribute__((ext_vector_type(8)))  float    v8f;
typedef __attribute__((ext_vector_type(4)))  float    f4;
typedef __attribute__((ext_vector_type(4)))  _Float16 h4;

#define N_NODES   65536
#define N_EDGES   1048576
#define N_GRAPHS  1024
#define N_PAIRS   512
#define DM        64      // hidden dim D
#define KX        192     // 3*D concat width
#define HMID      128     // MSG_H == UPD_H
#define GREP      128
#define N_PROP    5
#define KT1       6       // 192/32
#define NT1       8       // 128/16
#define KT2       4       // 128/32
#define NT2       4       // 64/16

// ---- WMMA helpers ---------------------------------------------------------

union Frag16 { v16h v; f4 q[2]; };

// A-fragment (16x32 f16, M rows striped per ISA 7.12.2) from an LDS row-major
// f16 buffer with `stride` halves per row. kcols = k-tile * 32 (column base).
// Per lane: two aligned 16B chunks: halves [kcols+half*8, +8) and [+16, +8).
__device__ inline v16h load_frag_A(const _Float16* base, int stride, int mt,
                                   int kcols, int lane) {
  int row  = mt * 16 + (lane & 15);
  int half = lane >> 4;
  const _Float16* p = base + row * stride + kcols + half * 8;
  Frag16 f;
  f.q[0] = *(const f4*)(p);
  f.q[1] = *(const f4*)(p + 16);
  return f.v;
}

// B-fragment from pre-packed weights: fragment `frag` is 32 lanes x 16 f16,
// contiguous per lane (two 16B global loads, fully coalesced across the wave).
__device__ inline v16h load_frag_B(const _Float16* pack, int frag, int lane) {
  const _Float16* p = pack + (((size_t)frag * 32) + lane) * 16;
  Frag16 f;
  f.q[0] = *(const f4*)(p);
  f.q[1] = *(const f4*)(p + 8);
  return f.v;
}

__device__ inline v8f wmma16(v16h a, v16h b, v8f c) {
  return __builtin_amdgcn_wmma_f32_16x16x32_f16(false, a, false, b,
                                                (short)0, c, false, false);
}

__device__ inline void store_h4(_Float16* dst, f4 a) {
  *(h4*)dst = __builtin_convertvector(a, h4);
}

// ---- Weight fragment packing ---------------------------------------------
// Pack W (K x Nw, row-major f32) into B-matrix lane image:
// out[((kt*NT+nt)*32 + lane)*16 + e] = W[(kt*32 + k_local)*Nw + nt*16 + lane%16]
// k_local(e, half) per the 16-bit operand layout (ISA 7.12.2).
__global__ void pack_w_kernel(const float* __restrict__ W,
                              _Float16* __restrict__ out, int K, int Nw) {
  int total = K * Nw;
  int i = blockIdx.x * blockDim.x + threadIdx.x;
  if (i >= total) return;
  int e    = i & 15;
  int l    = (i >> 4) & 31;
  int frag = i >> 9;
  int ntn  = Nw >> 4;
  int kt   = frag / ntn;
  int nt   = frag - kt * ntn;
  int half = l >> 4;
  int v    = e >> 1;
  int kl   = ((v & 4) ? 16 : 0) + half * 8 + ((v & 3) << 1) + (e & 1);
  int k    = kt * 32 + kl;
  int n    = nt * 16 + (l & 15);
  out[i] = (_Float16)W[(size_t)k * Nw + n];
}

// ---- Encoders -------------------------------------------------------------

__global__ void encode_nodes_kernel(const float* __restrict__ nf,
                                    const float* __restrict__ W,
                                    const float* __restrict__ b,
                                    float* __restrict__ h) {
  long gid = (long)blockIdx.x * blockDim.x + threadIdx.x;
  if (gid >= (long)N_NODES * DM) return;
  long n = gid >> 6; int c = (int)(gid & 63);
  float s = b[c];
  const float* x = nf + n * 32;
#pragma unroll
  for (int k = 0; k < 32; ++k) s += x[k] * W[k * DM + c];
  h[gid] = s;
}

__global__ void encode_edges_kernel(const float* __restrict__ ef,
                                    const float* __restrict__ W,
                                    const float* __restrict__ b,
                                    _Float16* __restrict__ e16) {
  long gid = (long)blockIdx.x * blockDim.x + threadIdx.x;
  if (gid >= (long)N_EDGES * DM) return;
  long n = gid >> 6; int c = (int)(gid & 63);
  float s = b[c];
  const float* x = ef + n * 16;
#pragma unroll
  for (int k = 0; k < 16; ++k) s += x[k] * W[k * DM + c];
  e16[gid] = (_Float16)s;
}

__global__ void fill_zero_kernel(float* __restrict__ p, long n) {
  long g = (long)blockIdx.x * blockDim.x + threadIdx.x;
  if (g < n) p[g] = 0.f;
}

// ---- Edge message MLPs (the hot kernel) -----------------------------------
// Block = 256 threads = 8 waves; tile = 128 edges. Wave w owns M-rows
// [w*16, w*16+16). LDS: X (128x192 f16), H1 (128x128 f16), edge indices.
// Pass 0: msgs = MLP([hf|ht|e]) scattered to agg[to]; pass 1: rmsgs =
// MLP([ht|hf|e]) scattered to agg[from] (column-block permutation of X).
__global__ __launch_bounds__(256)
void edge_mlp_kernel(const float* __restrict__ h,
                     const _Float16* __restrict__ e16,
                     const int* __restrict__ from_idx,
                     const int* __restrict__ to_idx,
                     const _Float16* __restrict__ w1m, const float* __restrict__ b1m,
                     const _Float16* __restrict__ w2m, const float* __restrict__ b2m,
                     const _Float16* __restrict__ w1r, const float* __restrict__ b1r,
                     const _Float16* __restrict__ w2r, const float* __restrict__ b2r,
                     float* __restrict__ agg) {
  extern __shared__ char smem[];
  _Float16* Xs  = (_Float16*)smem;          // 128*192 halves
  _Float16* H1s = Xs + 128 * KX;            // 128*128 halves
  int* toi   = (int*)(H1s + 128 * HMID);    // 128 ints
  int* fromi = toi + 128;                   // 128 ints

  const int tid  = threadIdx.x;
  const int lane = tid & 31;
  const int mt   = tid >> 5;
  const long e0  = (long)blockIdx.x * 128;

  // ---- gather: each thread fills half a row of [hf|ht|e] as f16 ----
  {
    int r   = tid & 127;
    int sel = tid >> 7;                 // column half: [0,32) or [32,64)
    long edge = e0 + r;
    int fi = from_idx[edge], ti = to_idx[edge];
    if (tid < 128) { fromi[r] = fi; toi[r] = ti; }
    const f4* hf = (const f4*)(h + (size_t)fi * DM + sel * 32);
    const f4* ht = (const f4*)(h + (size_t)ti * DM + sel * 32);
    _Float16* xr = Xs + r * KX;
#pragma unroll
    for (int j = 0; j < 8; ++j) {
      store_h4(xr +       sel * 32 + j * 4, hf[j]);
      store_h4(xr +  DM + sel * 32 + j * 4, ht[j]);
    }
    const f4* ep = (const f4*)(e16 + (size_t)edge * DM + sel * 32);
    f4* xo = (f4*)(xr + 2 * DM + sel * 32);
#pragma unroll
    for (int j = 0; j < 4; ++j) xo[j] = ep[j];
  }
  __syncthreads();

  for (int pass = 0; pass < 2; ++pass) {
    const _Float16* w1 = pass ? w1r : w1m;
    const float*    b1 = pass ? b1r : b1m;
    const _Float16* w2 = pass ? w2r : w2m;
    const float*    b2 = pass ? b2r : b2m;

    // layer 1: (16x192) @ (192x128), K in 6 steps of 32
    v8f acc[NT1];
#pragma unroll
    for (int nt = 0; nt < NT1; ++nt) acc[nt] = (v8f){0,0,0,0,0,0,0,0};
#pragma unroll
    for (int kt = 0; kt < KT1; ++kt) {
      // rmsg reads X with [ht|hf|e] column-block order
      int kc = (pass == 0) ? kt * 32
             : ((kt < 2) ? 64 + kt * 32 : (kt < 4) ? (kt - 2) * 32 : kt * 32);
      v16h a = load_frag_A(Xs, KX, mt, kc, lane);
#pragma unroll
      for (int nt = 0; nt < NT1; ++nt) {
        v16h b = load_frag_B(w1, kt * NT1 + nt, lane);
        acc[nt] = wmma16(a, b, acc[nt]);
      }
    }
    // bias + ReLU -> H1 (f16, this wave's own rows only)
#pragma unroll
    for (int nt = 0; nt < NT1; ++nt) {
      int col = nt * 16 + (lane & 15);
      float bias = b1[col];
#pragma unroll
      for (int r = 0; r < 8; ++r) {
        int row = mt * 16 + ((lane < 16) ? r : r + 8);
        float v = acc[nt][r] + bias;
        H1s[row * HMID + col] = (_Float16)fmaxf(v, 0.f);
      }
    }
    __syncthreads();

    // layer 2: (16x128) @ (128x64)
    v8f acc2[NT2];
#pragma unroll
    for (int nt = 0; nt < NT2; ++nt) acc2[nt] = (v8f){0,0,0,0,0,0,0,0};
#pragma unroll
    for (int kt = 0; kt < KT2; ++kt) {
      v16h a = load_frag_A(H1s, HMID, mt, kt * 32, lane);
#pragma unroll
      for (int nt = 0; nt < NT2; ++nt) {
        v16h b = load_frag_B(w2, kt * NT2 + nt, lane);
        acc2[nt] = wmma16(a, b, acc2[nt]);
      }
    }
    // segment-sum scatter (global f32 atomics into agg)
    const int* idx = pass ? fromi : toi;
#pragma unroll
    for (int nt = 0; nt < NT2; ++nt) {
      int col = nt * 16 + (lane & 15);
      float bias = b2[col];
#pragma unroll
      for (int r = 0; r < 8; ++r) {
        int row = mt * 16 + ((lane < 16) ? r : r + 8);
        atomicAdd(&agg[(size_t)idx[row] * DM + col], acc2[nt][r] + bias);
      }
    }
    __syncthreads();
  }
}

// ---- Cross-graph attention (per pair, fp32 VALU; tiny FLOPs) --------------
__global__ __launch_bounds__(256)
void attention_kernel(const float* __restrict__ h, float* __restrict__ att) {
  extern __shared__ char smem[];
  float* xs   = (float*)smem;      // 64x64
  float* ys   = xs + 4096;
  float* sbuf = ys + 4096;
  float* pbuf = sbuf + 4096;
  float* qbuf = pbuf + 4096;
  float* rowmax = qbuf + 4096;
  float* rowsum = rowmax + 64;
  float* colmax = rowsum + 64;
  float* colsum = colmax + 64;

  const int tid = threadIdx.x;
  const size_t base = (size_t)blockIdx.x * 128 * DM;
  for (int i = tid; i < 4096; i += 256) {
    xs[i] = h[base + i];
    ys[i] = h[base + 4096 + i];
  }
  __syncthreads();
#pragma unroll 4
  for (int t = 0; t < 16; ++t) {
    int idx = tid + t * 256;
    int i = idx >> 6, j = idx & 63;
    float s = 0.f;
    for (int k = 0; k < 64; ++k) s += xs[i * 64 + k] * ys[j * 64 + k];
    sbuf[idx] = s;
  }
  __syncthreads();
  if (tid < 64) {
    float m = -1e30f;
    for (int j = 0; j < 64; ++j) m = fmaxf(m, sbuf[tid * 64 + j]);
    float s = 0.f;
    for (int j = 0; j < 64; ++j) s += __expf(sbuf[tid * 64 + j] - m);
    rowmax[tid] = m; rowsum[tid] = s;
  } else if (tid < 128) {
    int c = tid - 64;
    float m = -1e30f;
    for (int i = 0; i < 64; ++i) m = fmaxf(m, sbuf[i * 64 + c]);
    float s = 0.f;
    for (int i = 0; i < 64; ++i) s += __expf(sbuf[i * 64 + c] - m);
    colmax[c] = m; colsum[c] = s;
  }
  __syncthreads();
#pragma unroll 4
  for (int t = 0; t < 16; ++t) {
    int idx = tid + t * 256;
    int i = idx >> 6, j = idx & 63;
    float s = sbuf[idx];
    pbuf[idx] = __expf(s - rowmax[i]) / rowsum[i];
    qbuf[idx] = __expf(s - colmax[j]) / colsum[j];
  }
  __syncthreads();
#pragma unroll 4
  for (int t = 0; t < 16; ++t) {
    int idx = tid + t * 256;
    int i = idx >> 6, d = idx & 63;
    float ax = 0.f, ay = 0.f;
    for (int k = 0; k < 64; ++k) {
      ax += pbuf[i * 64 + k] * ys[k * 64 + d];   // attn_x[i]
      ay += qbuf[k * 64 + i] * xs[k * 64 + d];   // attn_y[j==i]
    }
    att[base + i * 64 + d] = ax;
    att[base + 4096 + i * 64 + d] = ay;
  }
}

// ---- Node update MLP (WMMA, residual) -------------------------------------
__global__ __launch_bounds__(256)
void node_update_kernel(float* __restrict__ h,
                        const float* __restrict__ agg,
                        const float* __restrict__ att,
                        const _Float16* __restrict__ w1, const float* __restrict__ b1,
                        const _Float16* __restrict__ w2, const float* __restrict__ b2) {
  extern __shared__ char smem[];
  _Float16* Xs  = (_Float16*)smem;       // 128*192
  _Float16* H1s = Xs + 128 * KX;         // 128*128

  const int tid  = threadIdx.x;
  const int lane = tid & 31;
  const int mt   = tid >> 5;

  {
    int r   = tid & 127;
    int sel = tid >> 7;
    size_t node = (size_t)blockIdx.x * 128 + r;
    const f4* ag = (const f4*)(agg + node * DM + sel * 32);
    const f4* hh = (const f4*)(h   + node * DM + sel * 32);
    const f4* at = (const f4*)(att + node * DM + sel * 32);
    _Float16* xr = Xs + r * KX;
#pragma unroll
    for (int j = 0; j < 8; ++j) {
      f4 hv = hh[j], av = at[j];
      store_h4(xr +          sel * 32 + j * 4, ag[j]);        // agg
      store_h4(xr +     DM + sel * 32 + j * 4, hv - av);      // h - attended
      store_h4(xr + 2 * DM + sel * 32 + j * 4, hv);           // h
    }
  }
  __syncthreads();

  v8f acc[NT1];
#pragma unroll
  for (int nt = 0; nt < NT1; ++nt) acc[nt] = (v8f){0,0,0,0,0,0,0,0};
#pragma unroll
  for (int kt = 0; kt < KT1; ++kt) {
    v16h a = load_frag_A(Xs, KX, mt, kt * 32, lane);
#pragma unroll
    for (int nt = 0; nt < NT1; ++nt) {
      v16h b = load_frag_B(w1, kt * NT1 + nt, lane);
      acc[nt] = wmma16(a, b, acc[nt]);
    }
  }
#pragma unroll
  for (int nt = 0; nt < NT1; ++nt) {
    int col = nt * 16 + (lane & 15);
    float bias = b1[col];
#pragma unroll
    for (int r = 0; r < 8; ++r) {
      int row = mt * 16 + ((lane < 16) ? r : r + 8);
      float v = acc[nt][r] + bias;
      H1s[row * HMID + col] = (_Float16)fmaxf(v, 0.f);
    }
  }
  __syncthreads();

  v8f acc2[NT2];
#pragma unroll
  for (int nt = 0; nt < NT2; ++nt) acc2[nt] = (v8f){0,0,0,0,0,0,0,0};
#pragma unroll
  for (int kt = 0; kt < KT2; ++kt) {
    v16h a = load_frag_A(H1s, HMID, mt, kt * 32, lane);
#pragma unroll
    for (int nt = 0; nt < NT2; ++nt) {
      v16h b = load_frag_B(w2, kt * NT2 + nt, lane);
      acc2[nt] = wmma16(a, b, acc2[nt]);
    }
  }
#pragma unroll
  for (int nt = 0; nt < NT2; ++nt) {
    int col = nt * 16 + (lane & 15);
    float bias = b2[col];
#pragma unroll
    for (int r = 0; r < 8; ++r) {
      int row = mt * 16 + ((lane < 16) ? r : r + 8);
      size_t g = ((size_t)blockIdx.x * 128 + row) * DM + col;
      h[g] = h[g] + acc2[nt][r] + bias;   // residual
    }
  }
}

// ---- Graph readout: gated sum + linear ------------------------------------
__global__ __launch_bounds__(256)
void graph_agg_kernel(const float* __restrict__ h,
                      const float* __restrict__ W1, const float* __restrict__ b1,
                      const float* __restrict__ W2, const float* __restrict__ b2,
                      float* __restrict__ gs) {
  __shared__ float hs[4096];
  __shared__ float gbuf[256];
  __shared__ float garr[128];
  const int tid = threadIdx.x;
  const size_t g = blockIdx.x;
  for (int i = tid; i < 4096; i += 256) hs[i] = h[g * 4096 + i];
  __syncthreads();
  float acc = 0.f;
  for (int n = 0; n < 64; ++n) {
    float s = b1[tid];
    const float* hr = hs + n * 64;
    for (int k = 0; k < 64; ++k) s += hr[k] * W1[k * 256 + tid];
    gbuf[tid] = s;
    __syncthreads();
    if (tid < 128) {
      float a = gbuf[tid];
      acc += (1.f / (1.f + __expf(-a))) * gbuf[tid + 128];
    }
    __syncthreads();
  }
  if (tid < 128) garr[tid] = acc;
  __syncthreads();
  if (tid < 128) {
    float o = b2[tid];
    for (int k = 0; k < 128; ++k) o += garr[k] * W2[k * 128 + tid];
    gs[g * 128 + tid] = o;
  }
}

__global__ __launch_bounds__(128)
void score_kernel(const float* __restrict__ gs, float* __restrict__ out) {
  __shared__ float red[128];
  const int t = threadIdx.x;
  const size_t b = blockIdx.x;
  float va = gs[(2 * b) * 128 + t];
  float vb = gs[(2 * b + 1) * 128 + t];
  red[t] = fmaxf(va - vb, 0.f);
  __syncthreads();
  for (int s = 64; s > 0; s >>= 1) {
    if (t < s) red[t] += red[t + s];
    __syncthreads();
  }
  if (t == 0) out[b] = -red[0];
}

// ---------------------------------------------------------------------------

extern "C" void kernel_launch(void* const* d_in, const int* in_sizes, int n_in,
                              void* d_out, int out_size, void* d_ws, size_t ws_size,
                              hipStream_t stream) {
  const float* node_features = (const float*)d_in[0];
  const float* edge_features = (const float*)d_in[1];
  const int*   from_idx      = (const int*)d_in[2];
  const int*   to_idx        = (const int*)d_in[3];
  const float* enc_node_W = (const float*)d_in[4];
  const float* enc_node_b = (const float*)d_in[5];
  const float* enc_edge_W = (const float*)d_in[6];
  const float* enc_edge_b = (const float*)d_in[7];
  const float* msg_W1  = (const float*)d_in[8];
  const float* msg_b1  = (const float*)d_in[9];
  const float* msg_W2  = (const float*)d_in[10];
  const float* msg_b2  = (const float*)d_in[11];
  const float* rmsg_W1 = (const float*)d_in[12];
  const float* rmsg_b1 = (const float*)d_in[13];
  const float* rmsg_W2 = (const float*)d_in[14];
  const float* rmsg_b2 = (const float*)d_in[15];
  const float* upd_W1  = (const float*)d_in[16];
  const float* upd_b1  = (const float*)d_in[17];
  const float* upd_W2  = (const float*)d_in[18];
  const float* upd_b2  = (const float*)d_in[19];
  const float* agg_W1  = (const float*)d_in[20];
  const float* agg_b1  = (const float*)d_in[21];
  const float* agg_W2  = (const float*)d_in[22];
  const float* agg_b2  = (const float*)d_in[23];

  char* wsb = (char*)d_ws;
  const size_t OFF_H   = 0;
  const size_t OFF_AGG = 16ull << 20;
  const size_t OFF_ATT = 32ull << 20;
  const size_t OFF_E16 = 48ull << 20;
  const size_t OFF_PK  = 176ull << 20;
  const size_t OFF_GS  = (176ull << 20) + 256 * 1024;

  float*    h    = (float*)(wsb + OFF_H);
  float*    agg  = (float*)(wsb + OFF_AGG);
  float*    att  = (float*)(wsb + OFF_ATT);
  _Float16* e16  = (_Float16*)(wsb + OFF_E16);
  _Float16* pk   = (_Float16*)(wsb + OFF_PK);
  _Float16* w1m  = pk;                  // 6*8*512  = 24576 halves
  _Float16* w2m  = pk + 24576;          // 4*4*512  =  8192
  _Float16* w1r  = pk + 32768;
  _Float16* w2r  = pk + 57344;
  _Float16* w1u  = pk + 65536;
  _Float16* w2u  = pk + 90112;
  float*    gs   = (float*)(wsb + OFF_GS);
  float*    out  = (float*)d_out;

  // weight fragment packing (tiny)
  pack_w_kernel<<<(KX * HMID + 255) / 256, 256, 0, stream>>>(msg_W1,  w1m, KX, HMID);
  pack_w_kernel<<<(HMID * DM + 255) / 256, 256, 0, stream>>>(msg_W2,  w2m, HMID, DM);
  pack_w_kernel<<<(KX * HMID + 255) / 256, 256, 0, stream>>>(rmsg_W1, w1r, KX, HMID);
  pack_w_kernel<<<(HMID * DM + 255) / 256, 256, 0, stream>>>(rmsg_W2, w2r, HMID, DM);
  pack_w_kernel<<<(KX * HMID + 255) / 256, 256, 0, stream>>>(upd_W1,  w1u, KX, HMID);
  pack_w_kernel<<<(HMID * DM + 255) / 256, 256, 0, stream>>>(upd_W2,  w2u, HMID, DM);

  // encoders
  encode_nodes_kernel<<<(N_NODES * DM) / 256, 256, 0, stream>>>(
      node_features, enc_node_W, enc_node_b, h);
  encode_edges_kernel<<<(int)(((long)N_EDGES * DM) / 256), 256, 0, stream>>>(
      edge_features, enc_edge_W, enc_edge_b, e16);

  const size_t SMEM_EDGE = 128 * KX * 2 + 128 * HMID * 2 + 256 * 4; // 82944 B
  const size_t SMEM_UPD  = 128 * KX * 2 + 128 * HMID * 2;          // 81920 B
  const size_t SMEM_ATT  = (5 * 4096 + 256) * 4;                   // 82944 B

  for (int it = 0; it < N_PROP; ++it) {
    fill_zero_kernel<<<(N_NODES * DM) / 256, 256, 0, stream>>>(agg,
                                                               (long)N_NODES * DM);
    edge_mlp_kernel<<<N_EDGES / 128, 256, SMEM_EDGE, stream>>>(
        h, e16, from_idx, to_idx,
        w1m, msg_b1, w2m, msg_b2,
        w1r, rmsg_b1, w2r, rmsg_b2, agg);
    attention_kernel<<<N_PAIRS, 256, SMEM_ATT, stream>>>(h, att);
    node_update_kernel<<<N_NODES / 128, 256, SMEM_UPD, stream>>>(
        h, agg, att, w1u, upd_b1, w2u, upd_b2);
  }

  graph_agg_kernel<<<N_GRAPHS, 256, 0, stream>>>(h, agg_W1, agg_b1,
                                                 agg_W2, agg_b2, gs);
  score_kernel<<<N_PAIRS, 128, 0, stream>>>(gs, out);
}
